// FactorGraphGRU_26216480375069
// MI455X (gfx1250) — compile-verified, matrix operand order checked by reading, not run on hardware
//
#include <hip/hip_runtime.h>

// ---------------------------------------------------------------------------
// FactorGraphGRU on MI455X (gfx1250): all large GEMMs via v_wmma_f32_16x16x32_f16
// N=4096, H=256, HEADS=3. Adjacencies (2x64MB) are L2-resident (192MB L2).
// GEMM: 128x64 block tile, 8 waves, 32x32 per wave (4 WMMA/k-step),
// LDS double-buffered with register prefetch (1 barrier per k-step).
// ---------------------------------------------------------------------------

typedef __attribute__((ext_vector_type(16))) _Float16 v16h;
typedef __attribute__((ext_vector_type(8)))  _Float16 v8h;
typedef __attribute__((ext_vector_type(8)))  float    v8f;
typedef __attribute__((ext_vector_type(4)))  float    v4f;

#define NN   4096
#define HH   256
#define NH   768   // 3*H
#define BM   128
#define BN   64
#define BK   32
#define LDT  40    // padded LDS row stride in halves (multiple of 8 -> 16B aligned)

// --------------------------- small prep kernels ----------------------------

__global__ void fg_diag(const float* __restrict__ na, const float* __restrict__ ea,
                        float* __restrict__ dn, float* __restrict__ de, int n) {
  int i = blockIdx.x * blockDim.x + threadIdx.x;
  if (i < n) {
    dn[i] = na[(size_t)i * n + i];
    de[i] = ea[(size_t)i * n + i];
  }
}

__global__ void fg_cvt16(const float* __restrict__ in, _Float16* __restrict__ out, int n) {
  int i = blockIdx.x * blockDim.x + threadIdx.x;
  if (i < n) out[i] = (_Float16)in[i];
}

// out[c*rows + r] = (f16) in[r*cols + c]
__global__ void fg_tr16(const float* __restrict__ in, _Float16* __restrict__ out,
                        int rows, int cols) {
  int i = blockIdx.x * blockDim.x + threadIdx.x;
  if (i < rows * cols) {
    int r = i / cols, c = i % cols;
    out[(size_t)c * rows + r] = (_Float16)in[i];
  }
}

// --------------------------- WMMA fragment helpers -------------------------

union FragU  { v16h v; v8h h8[2]; };
union PackU  { v8h h8[2]; _Float16 f[16]; };

__device__ __forceinline__ v16h fg_ldfrag(const _Float16* base, int fr, int kb) {
  // ISA 16-bit A layout: lane<16 -> K {0..7, 16..23}, lane>=16 -> K {8..15, 24..31}
  FragU u;
  const _Float16* p = base + fr * LDT + kb;
  u.h8[0] = *(const v8h*)p;
  u.h8[1] = *(const v8h*)(p + 16);
  return u.v;
}

#define FG_WMMA(acc, a, b) \
  acc = __builtin_amdgcn_wmma_f32_16x16x32_f16(false, a, false, b, (short)0, acc, false, false)

// ------------------------ generic f16 GEMM (WMMA) --------------------------
// C[M,Nc] = A[M,K] * Bt[Nc,K]^T (+ bias[Nc]); A,Bt f16 row-major, C f32.
__global__ void __launch_bounds__(256)
fg_gemm16(const _Float16* __restrict__ A, const _Float16* __restrict__ Bt,
          float* __restrict__ C, const float* __restrict__ bias,
          int M, int Nc, int K) {
  __shared__ _Float16 As[2][BM * LDT];
  __shared__ _Float16 Bs[2][BN * LDT];
  const int t    = threadIdx.x;
  const int lane = t & 31;
  const int w    = t >> 5;
  const int wr   = w & 3;      // 32-row slab 0..3
  const int cw   = w >> 2;     // 32-col slab 0..1
  const int row0 = blockIdx.y * BM;
  const int col0 = blockIdx.x * BN;
  const int ar   = t >> 1;          // A staging: 128 rows x 32 halves, 16/thread
  const int ac   = (t & 1) * 16;
  const int br   = t >> 2;          // B staging: 64 rows x 32 halves, 8/thread
  const int bc   = (t & 3) * 8;
  const int fr   = lane & 15;
  const int kb   = (lane >> 4) * 8;

  const _Float16* Aptr = A  + (size_t)(row0 + ar) * K + ac;
  const _Float16* Bptr = Bt + (size_t)(col0 + br) * K + bc;

  // prologue: stage tile 0 into buffer 0
  *(v8h*)&As[0][ar * LDT + ac]     = *(const v8h*)(Aptr);
  *(v8h*)&As[0][ar * LDT + ac + 8] = *(const v8h*)(Aptr + 8);
  *(v8h*)&Bs[0][br * LDT + bc]     = *(const v8h*)(Bptr);
  __syncthreads();

  v8f acc00 = {}, acc01 = {}, acc10 = {}, acc11 = {};
  const int nk = K / BK;
  for (int ks = 0; ks < nk; ++ks) {
    const int cur = ks & 1, nxt = cur ^ 1;
    v8h pa0, pa1, pb;
    if (ks + 1 < nk) {                       // register prefetch of next tile
      const _Float16* An = Aptr + (size_t)(ks + 1) * BK;
      const _Float16* Bn = Bptr + (size_t)(ks + 1) * BK;
      pa0 = *(const v8h*)An;
      pa1 = *(const v8h*)(An + 8);
      pb  = *(const v8h*)Bn;
    }
    v16h a0 = fg_ldfrag(&As[cur][(wr * 32) * LDT],      fr, kb);
    v16h a1 = fg_ldfrag(&As[cur][(wr * 32 + 16) * LDT], fr, kb);
    v16h b0 = fg_ldfrag(&Bs[cur][(cw * 32) * LDT],      fr, kb);
    v16h b1 = fg_ldfrag(&Bs[cur][(cw * 32 + 16) * LDT], fr, kb);
    FG_WMMA(acc00, a0, b0);
    FG_WMMA(acc01, a0, b1);
    FG_WMMA(acc10, a1, b0);
    FG_WMMA(acc11, a1, b1);
    if (ks + 1 < nk) {
      *(v8h*)&As[nxt][ar * LDT + ac]     = pa0;
      *(v8h*)&As[nxt][ar * LDT + ac + 8] = pa1;
      *(v8h*)&Bs[nxt][br * LDT + bc]     = pb;
    }
    __syncthreads();
  }

  // C layout: lanes 0-15 hold M 0..7, lanes 16-31 hold M 8..15, N = lane&15
  const int n   = lane & 15;
  const int mb  = (lane >> 4) * 8;
  const int gr0 = row0 + wr * 32 + mb;
  const int gc  = col0 + cw * 32 + n;
  const float bv0 = bias ? bias[gc]      : 0.f;
  const float bv1 = bias ? bias[gc + 16] : 0.f;
#pragma unroll
  for (int v = 0; v < 8; ++v) {
    C[(size_t)(gr0 + v) * Nc + gc]           = acc00[v] + bv0;
    C[(size_t)(gr0 + v) * Nc + gc + 16]      = acc01[v] + bv1;
    C[(size_t)(gr0 + 16 + v) * Nc + gc]      = acc10[v] + bv0;
    C[(size_t)(gr0 + 16 + v) * Nc + gc + 16] = acc11[v] + bv1;
  }
}

// ---------------- adjacency-transform GEMM (WMMA, f32 A source) ------------
// mode 0: A = adj with zeroed diagonal (values {-1,0,1})
// mode 1: A = (adj>0), diag zeroed     mode 2: A = (adj<0), diag zeroed
// mode 3: A = sign(adj)*exp(score_j)*rowscale_r per head (blockIdx.z = head)
struct AdjPre { v4f x0, x1, x2, x3; };

__device__ __forceinline__ void fg_adj_xform(
    _Float16* dst, const AdjPre& p, int gr, int k0, int ac, int mode,
    const float* EPh, const float* EMh, float rsv) {
  PackU o;
#pragma unroll
  for (int e = 0; e < 16; ++e) {
    float v = (e < 4) ? p.x0[e] : (e < 8) ? p.x1[e - 4] : (e < 12) ? p.x2[e - 8] : p.x3[e - 12];
    int   j = k0 + ac + e;
    float wv = 0.f;
    if (mode == 0)      wv = (gr == j) ? 0.f : v;
    else if (mode == 1) wv = (gr != j && v > 0.f) ? 1.f : 0.f;
    else if (mode == 2) wv = (gr != j && v < 0.f) ? 1.f : 0.f;
    else if (gr != j && v != 0.f) {
      float ev = (v > 0.f) ? EPh[j] : EMh[j];
      wv = ((v > 0.f) ? ev : -ev) * rsv;   // normalized softmax weight * sign, |wv|<=1
    }
    o.f[e] = (_Float16)wv;
  }
  *(v8h*)dst       = o.h8[0];
  *(v8h*)(dst + 8) = o.h8[1];
}

__global__ void __launch_bounds__(256)
fg_adjgemm(const float* __restrict__ Adj, const _Float16* __restrict__ Bt,
           float* __restrict__ C, int M, int Nc, int K, int mode,
           const float* __restrict__ EP, const float* __restrict__ EM,
           const float* __restrict__ RS) {
  __shared__ _Float16 As[2][BM * LDT];
  __shared__ _Float16 Bs[2][BN * LDT];
  const int t    = threadIdx.x;
  const int lane = t & 31;
  const int w    = t >> 5;
  const int wr   = w & 3;
  const int cw   = w >> 2;
  const int row0 = blockIdx.y * BM;
  const int col0 = blockIdx.x * BN;
  const int head = blockIdx.z;
  const int ar   = t >> 1;
  const int ac   = (t & 1) * 16;
  const int br   = t >> 2;
  const int bc   = (t & 3) * 8;
  const int fr   = lane & 15;
  const int kb   = (lane >> 4) * 8;

  const _Float16* BtH = Bt + (mode == 3 ? (size_t)head * HH * K : 0);
  float*          CH  = C  + (mode == 3 ? (size_t)head * (size_t)M * Nc : 0);
  const float* EPh = (mode == 3) ? EP + (size_t)head * K : nullptr;
  const float* EMh = (mode == 3) ? EM + (size_t)head * K : nullptr;
  const int   gr   = row0 + ar;
  const float rsv  = (mode == 3) ? RS[(size_t)head * M + gr] : 0.f;
  const float* arow = Adj + (size_t)gr * K + ac;
  const _Float16* Bptr = BtH + (size_t)(col0 + br) * K + bc;

  // prologue: stage tile 0 into buffer 0
  {
    AdjPre p;
    p.x0 = *(const v4f*)(arow);
    p.x1 = *(const v4f*)(arow + 4);
    p.x2 = *(const v4f*)(arow + 8);
    p.x3 = *(const v4f*)(arow + 12);
    fg_adj_xform(&As[0][ar * LDT + ac], p, gr, 0, ac, mode, EPh, EMh, rsv);
    *(v8h*)&Bs[0][br * LDT + bc] = *(const v8h*)(Bptr);
  }
  __syncthreads();

  v8f acc00 = {}, acc01 = {}, acc10 = {}, acc11 = {};
  const int nk = K / BK;
  for (int ks = 0; ks < nk; ++ks) {
    const int cur = ks & 1, nxt = cur ^ 1;
    AdjPre p;
    v8h pb;
    if (ks + 1 < nk) {                       // register prefetch of next tile
      const float* an = arow + (size_t)(ks + 1) * BK;
      p.x0 = *(const v4f*)(an);
      p.x1 = *(const v4f*)(an + 4);
      p.x2 = *(const v4f*)(an + 8);
      p.x3 = *(const v4f*)(an + 12);
      pb   = *(const v8h*)(Bptr + (size_t)(ks + 1) * BK);
    }
    v16h a0 = fg_ldfrag(&As[cur][(wr * 32) * LDT],      fr, kb);
    v16h a1 = fg_ldfrag(&As[cur][(wr * 32 + 16) * LDT], fr, kb);
    v16h b0 = fg_ldfrag(&Bs[cur][(cw * 32) * LDT],      fr, kb);
    v16h b1 = fg_ldfrag(&Bs[cur][(cw * 32 + 16) * LDT], fr, kb);
    FG_WMMA(acc00, a0, b0);
    FG_WMMA(acc01, a0, b1);
    FG_WMMA(acc10, a1, b0);
    FG_WMMA(acc11, a1, b1);
    if (ks + 1 < nk) {
      fg_adj_xform(&As[nxt][ar * LDT + ac], p, gr, (ks + 1) * BK, ac, mode, EPh, EMh, rsv);
      *(v8h*)&Bs[nxt][br * LDT + bc] = pb;
    }
    __syncthreads();
  }

  const int n   = lane & 15;
  const int mb  = (lane >> 4) * 8;
  const int gr0 = row0 + wr * 32 + mb;
  const int gc  = col0 + cw * 32 + n;
#pragma unroll
  for (int v = 0; v < 8; ++v) {
    CH[(size_t)(gr0 + v) * Nc + gc]           = acc00[v];
    CH[(size_t)(gr0 + v) * Nc + gc + 16]      = acc01[v];
    CH[(size_t)(gr0 + 16 + v) * Nc + gc]      = acc10[v];
    CH[(size_t)(gr0 + 16 + v) * Nc + gc + 16] = acc11[v];
  }
}

// ------------------------------ GAT small kernels --------------------------

// scores per (head, node): e_plus/e_minus with leaky_relu; also exp'd values
__global__ void fg_score(const float* __restrict__ hp, const float* __restrict__ hm,
                         const float* __restrict__ a,
                         float* __restrict__ epr, float* __restrict__ emr,
                         float* __restrict__ epe, float* __restrict__ eme) {
  int idx = blockIdx.x * blockDim.x + threadIdx.x;
  if (idx >= 3 * NN) return;
  int hd = idx / NN, n = idx % NN;
  const float* a0 = a + hd * 2 * HH;
  const float* a1 = a0 + HH;
  size_t base = (size_t)n * NH + hd * HH;
  float sp = 0.f, sm = 0.f;
  for (int c = 0; c < HH; ++c) {
    float p = hp[base + c], m = hm[base + c];
    sp += p * a0[c] + m * a1[c];
    sm += m * a0[c] + p * a1[c];
  }
  sp = (sp > 0.f) ? sp : 0.2f * sp;
  sm = (sm > 0.f) ? sm : 0.2f * sm;
  epr[idx] = sp; emr[idx] = sm;
  epe[idx] = expf(sp); eme[idx] = expf(sm);
}

// per-row softmax stats over edge_adj (diag excluded): rowscale = exp(-max)/denom
__global__ void __launch_bounds__(256)
fg_rowstats(const float* __restrict__ adj, const float* __restrict__ epr,
            const float* __restrict__ emr, float* __restrict__ rs) {
  const int r = blockIdx.x;
  const int t = threadIdx.x;
  __shared__ float red[256];
  float m[3] = {-3.0e38f, -3.0e38f, -3.0e38f};
  for (int j = t; j < NN; j += 256) {
    float v = adj[(size_t)r * NN + j];
    if (j == r || v == 0.f) continue;
    for (int hd = 0; hd < 3; ++hd) {
      float s = (v > 0.f) ? epr[hd * NN + j] : emr[hd * NN + j];
      m[hd] = fmaxf(m[hd], s);
    }
  }
  for (int hd = 0; hd < 3; ++hd) {
    red[t] = m[hd]; __syncthreads();
    for (int o = 128; o > 0; o >>= 1) { if (t < o) red[t] = fmaxf(red[t], red[t + o]); __syncthreads(); }
    m[hd] = red[0]; __syncthreads();
  }
  float d[3] = {0.f, 0.f, 0.f};
  for (int j = t; j < NN; j += 256) {
    float v = adj[(size_t)r * NN + j];
    if (j == r || v == 0.f) continue;
    for (int hd = 0; hd < 3; ++hd) {
      float s = (v > 0.f) ? epr[hd * NN + j] : emr[hd * NN + j];
      d[hd] += expf(s - m[hd]);
    }
  }
  for (int hd = 0; hd < 3; ++hd) {
    red[t] = d[hd]; __syncthreads();
    for (int o = 128; o > 0; o >>= 1) { if (t < o) red[t] += red[t + o]; __syncthreads(); }
    d[hd] = red[0]; __syncthreads();
  }
  if (t == 0)
    for (int hd = 0; hd < 3; ++hd)
      rs[hd * NN + r] = (d[hd] > 0.f && m[hd] > -1.0e38f) ? expf(-m[hd]) / d[hd] : 0.f;
}

// edge_support = mean over heads of attention outputs, plus f16 copy
__global__ void fg_combine(const float* __restrict__ Y, float* __restrict__ es,
                           _Float16* __restrict__ es16) {
  int i = blockIdx.x * blockDim.x + threadIdx.x;
  if (i >= NN * HH) return;
  const size_t S = (size_t)NN * HH;
  float s = (Y[i] + Y[S + i] + Y[2 * S + i]) * (1.f / 3.f);
  es[i] = s; es16[i] = (_Float16)s;
}

// GRUCell elementwise: gates [r,z,n] in 768-wide gi/gh
__global__ void fg_gru(const float* __restrict__ gi, const float* __restrict__ gh,
                       const float* __restrict__ hs, float* __restrict__ out) {
  int i = blockIdx.x * blockDim.x + threadIdx.x;
  if (i >= NN * HH) return;
  int r = i / HH, c = i % HH;
  size_t b = (size_t)r * NH;
  float ir = gi[b + c], iz = gi[b + HH + c], in_ = gi[b + 2 * HH + c];
  float hr = gh[b + c], hz = gh[b + HH + c], hn  = gh[b + 2 * HH + c];
  float rg = 1.f / (1.f + expf(-(ir + hr)));
  float zg = 1.f / (1.f + expf(-(iz + hz)));
  float ng = tanhf(in_ + rg * hn);
  out[i] = (1.f - zg) * ng + zg * hs[i];
}

__global__ void fg_final(const float* __restrict__ eo, const float* __restrict__ no,
                         const float* __restrict__ dn, const float* __restrict__ de,
                         float* __restrict__ out) {
  int i = blockIdx.x * blockDim.x + threadIdx.x;
  if (i >= NN * HH) return;
  int r = i / HH;
  out[i] = de[r] * eo[i] + dn[r] * no[i];
}

// ------------------------------- host driver -------------------------------

extern "C" void kernel_launch(void* const* d_in, const int* in_sizes, int n_in,
                              void* d_out, int out_size, void* d_ws, size_t ws_size,
                              hipStream_t stream) {
  const float* h        = (const float*)d_in[0];
  const float* node_adj = (const float*)d_in[1];
  const float* edge_adj = (const float*)d_in[2];
  const float* gat_W    = (const float*)d_in[3];
  const float* gat_a    = (const float*)d_in[4];
  const float* eWih     = (const float*)d_in[5];
  const float* eWhh     = (const float*)d_in[6];
  const float* ebih     = (const float*)d_in[7];
  const float* ebhh     = (const float*)d_in[8];
  const float* nWih     = (const float*)d_in[9];
  const float* nWhh     = (const float*)d_in[10];
  const float* nbih     = (const float*)d_in[11];
  const float* nbhh     = (const float*)d_in[12];
  float* out = (float*)d_out;

  char* ws = (char*)d_ws;
  size_t off = 0;
  auto alloc = [&](size_t bytes) -> void* {
    void* p = ws + off;
    off = (off + bytes + 255) & ~(size_t)255;
    return p;
  };
  const size_t NHs = (size_t)NN * HH;   // 1M
  const size_t NNH = (size_t)NN * NH;   // 3M

  _Float16* h16    = (_Float16*)alloc(NHs * 2);          // h  f16 row-major
  _Float16* hT16   = (_Float16*)alloc(NHs * 2);          // h^T [256][4096]
  _Float16* ns16   = (_Float16*)alloc(NHs * 2);
  _Float16* es16   = (_Float16*)alloc(NHs * 2);
  _Float16* hWT16  = (_Float16*)alloc(NNH * 2);          // [768][4096]
  _Float16* WcatT  = (_Float16*)alloc((size_t)NH * HH * 2);
  _Float16* eWih16 = (_Float16*)alloc((size_t)NH * HH * 2);
  _Float16* eWhh16 = (_Float16*)alloc((size_t)NH * HH * 2);
  _Float16* nWih16 = (_Float16*)alloc((size_t)NH * HH * 2);
  _Float16* nWhh16 = (_Float16*)alloc((size_t)NH * HH * 2);
  float* dn  = (float*)alloc(NN * 4);
  float* de  = (float*)alloc(NN * 4);
  float* ns  = (float*)alloc(NHs * 4);                   // node_support
  float* X1  = (float*)alloc(NNH * 4);                   // gi / hp / gi2
  float* X2  = (float*)alloc(NNH * 4);                   // gh / hm / gh2
  float* Y   = (float*)alloc(NNH * 4);                   // hW f32, then att out [3][N][H]
  float* eo  = (float*)alloc(NHs * 4);                   // edge_output
  float* no  = (float*)alloc(NHs * 4);                   // node_output
  float* es  = (float*)alloc(NHs * 4);                   // edge_support
  float* epr = (float*)alloc(3 * NN * 4);
  float* emr = (float*)alloc(3 * NN * 4);
  float* epe = (float*)alloc(3 * NN * 4);
  float* eme = (float*)alloc(3 * NN * 4);
  float* rsc = (float*)alloc(3 * NN * 4);
  (void)ws_size; (void)in_sizes; (void)n_in; (void)out_size;

  auto blks = [](size_t n) { return dim3((unsigned)((n + 255) / 256)); };
  const dim3 T256(256);

  // --- prep ---
  fg_diag<<<blks(NN), T256, 0, stream>>>(node_adj, edge_adj, dn, de, NN);
  fg_cvt16<<<blks(NHs), T256, 0, stream>>>(h, h16, (int)NHs);
  fg_tr16<<<blks(NHs), T256, 0, stream>>>(h, hT16, NN, HH);
  for (int hd = 0; hd < 3; ++hd)
    fg_tr16<<<blks((size_t)HH * HH), T256, 0, stream>>>(gat_W + (size_t)hd * HH * HH,
                                                        WcatT + (size_t)hd * HH * HH, HH, HH);
  fg_cvt16<<<blks((size_t)NH * HH), T256, 0, stream>>>(eWih, eWih16, NH * HH);
  fg_cvt16<<<blks((size_t)NH * HH), T256, 0, stream>>>(eWhh, eWhh16, NH * HH);
  fg_cvt16<<<blks((size_t)NH * HH), T256, 0, stream>>>(nWih, nWih16, NH * HH);
  fg_cvt16<<<blks((size_t)NH * HH), T256, 0, stream>>>(nWhh, nWhh16, NH * HH);

  // --- node_support = node_adj_norm @ h ---
  fg_adjgemm<<<dim3(HH / BN, NN / BM, 1), T256, 0, stream>>>(
      node_adj, hT16, ns, NN, HH, NN, 0, nullptr, nullptr, nullptr);
  fg_cvt16<<<blks(NHs), T256, 0, stream>>>(ns, ns16, (int)NHs);

  // --- edge GRU: gi = ns @ eWih^T + bih ; gh = h @ eWhh^T + bhh ---
  fg_gemm16<<<dim3(NH / BN, NN / BM), T256, 0, stream>>>(ns16, eWih16, X1, ebih, NN, NH, HH);
  fg_gemm16<<<dim3(NH / BN, NN / BM), T256, 0, stream>>>(h16, eWhh16, X2, ebhh, NN, NH, HH);
  fg_gru<<<blks(NHs), T256, 0, stream>>>(X1, X2, h, eo);

  // --- hW for all 3 heads: [4096,768] = h @ Wcat ; then transpose to f16 ---
  fg_gemm16<<<dim3(NH / BN, NN / BM), T256, 0, stream>>>(h16, WcatT, Y, nullptr, NN, NH, HH);
  fg_tr16<<<blks(NNH), T256, 0, stream>>>(Y, hWT16, NN, NH);

  // --- hp = (adj>0)@hW, hm = (adj<0)@hW (all heads at once, 768 wide) ---
  fg_adjgemm<<<dim3(NH / BN, NN / BM, 1), T256, 0, stream>>>(
      node_adj, hWT16, X1, NN, NH, NN, 1, nullptr, nullptr, nullptr);
  fg_adjgemm<<<dim3(NH / BN, NN / BM, 1), T256, 0, stream>>>(
      node_adj, hWT16, X2, NN, NH, NN, 2, nullptr, nullptr, nullptr);

  // --- scores + row softmax stats ---
  fg_score<<<blks(3 * (size_t)NN), T256, 0, stream>>>(X1, X2, gat_a, epr, emr, epe, eme);
  fg_rowstats<<<dim3(NN), T256, 0, stream>>>(edge_adj, epr, emr, rsc);

  // --- attention aggregate per head: Y[hd] = (att*edge_adj) @ hW_hd ---
  fg_adjgemm<<<dim3(HH / BN, NN / BM, 3), T256, 0, stream>>>(
      edge_adj, hWT16, Y, NN, HH, NN, 3, epe, eme, rsc);

  // --- edge_support = mean(heads) ---
  fg_combine<<<blks(NHs), T256, 0, stream>>>(Y, es, es16);

  // --- node GRU ---
  fg_gemm16<<<dim3(NH / BN, NN / BM), T256, 0, stream>>>(es16, nWih16, X1, nbih, NN, NH, HH);
  fg_gemm16<<<dim3(NH / BN, NN / BM), T256, 0, stream>>>(h16, nWhh16, X2, nbhh, NN, NH, HH);
  fg_gru<<<blks(NHs), T256, 0, stream>>>(X1, X2, h, no);

  // --- out = de*edge_output + dn*node_output ---
  fg_final<<<blks(NHs), T256, 0, stream>>>(eo, no, dn, de, out);
}
